// Speller_87978110091403
// MI455X (gfx1250) — matrix operand order repaired and checked by reading, not compiled, not run on hardware
//
#include <hip/hip_runtime.h>
#include <hip/hip_bf16.h>
#include <cstdint>
#include <cstddef>

// ---------------------------------------------------------------------------
// Problem dimensions (match reference)
// ---------------------------------------------------------------------------
constexpr int B_   = 64;
constexpr int T_   = 1024;
constexpr int H_   = 512;
constexpr int CTX_ = 512;
constexpr int V_   = 1024;
constexpr int L_   = 400;
constexpr int HC_  = H_ + CTX_;   // 1024
constexpr int G_   = 4 * H_;      // 2048 (LSTM gate width)
constexpr int LOUT_ = L_ - 1;     // 399

// ---------------------------------------------------------------------------
// bf16 helpers + WMMA vector types (trivial types -> union-safe)
// ---------------------------------------------------------------------------
typedef __attribute__((ext_vector_type(16))) __bf16 v16bf;
typedef __attribute__((ext_vector_type(8)))  float  v8f;
typedef unsigned int u32x4 __attribute__((ext_vector_type(4)));

union ABvec { v16bf v; u32x4 q[2]; };

__device__ __forceinline__ float bf2f(unsigned short u) {
  return __uint_as_float(((unsigned)u) << 16);
}
__device__ __forceinline__ unsigned short f2bf(float f) {
  unsigned u = __float_as_uint(f);
  u += 0x7fffu + ((u >> 16) & 1u);              // round-to-nearest-even
  return (unsigned short)(u >> 16);
}

// ---------------------------------------------------------------------------
// Generic bf16 WMMA GEMM:  C(M,N) = A(M,K) @ W(N,K)^T + bias, optional ReLU.
// One 16x16 output tile per wave; K-loop in steps of 32 using
// v_wmma_f32_16x16x32_bf16. Lane->element mapping follows CDNA5 ISA 7.12.2:
//   A (16-bit, 16x32): lane m (0..15) half hi: VGPR0-3 = K[hi*8 .. hi*8+7],
//                      VGPR4-7 = K[16+hi*8 .. 16+hi*8+7]   (two b128 loads)
//   B (16-bit, 32x16): lane n half hi: same K order as A (two b128)
//   C (f32, 16x16):    lane n half hi: VGPR r -> row m = hi*8 + r
// Writes f32 (Cf) and/or bf16 (Cbf) outputs with row stride ldc.
// ---------------------------------------------------------------------------
__global__ void gemm_bf16_wmma(const unsigned short* __restrict__ A, int lda,
                               const unsigned short* __restrict__ W, int ldw,
                               const float* __restrict__ bias,
                               float* __restrict__ Cf,
                               unsigned short* __restrict__ Cbf, int ldc,
                               int M, int N, int K, int relu)
{
  const int wave   = (int)((blockIdx.x * blockDim.x + threadIdx.x) >> 5);
  const int lane   = (int)(threadIdx.x & 31);
  const int tilesN = N >> 4;
  const int total  = (M >> 4) * tilesN;
  if (wave >= total) return;                    // wave-uniform: EXEC stays all-1s

  const int mBase = (wave / tilesN) << 4;
  const int nBase = (wave % tilesN) << 4;
  const int tl = lane & 15;
  const int hi = lane >> 4;

  const unsigned short* Arow = A + (size_t)(mBase + tl) * (size_t)lda;
  const unsigned short* Wrow = W + (size_t)(nBase + tl) * (size_t)ldw;

  v8f acc = {};
  for (int kb = 0; kb < K; kb += 32) {
    ABvec a, b;
    a.q[0] = *(const u32x4*)(Arow + kb + hi * 8);
    a.q[1] = *(const u32x4*)(Arow + kb + 16 + hi * 8);
    b.q[0] = *(const u32x4*)(Wrow + kb + hi * 8);
    b.q[1] = *(const u32x4*)(Wrow + kb + 16 + hi * 8);
    acc = __builtin_amdgcn_wmma_f32_16x16x32_bf16(
        /*neg_a=*/false, a.v, /*neg_b=*/false, b.v,
        /*c_mod=*/(short)0, acc, /*reuse_a=*/false, /*reuse_b=*/false);
  }

  const float bn  = bias ? bias[nBase + tl] : 0.0f;
  const int   m0  = mBase + hi * 8;
  const int   col = nBase + tl;
#pragma unroll
  for (int r = 0; r < 8; ++r) {
    float v = acc[r] + bn;
    if (relu && v < 0.0f) v = 0.0f;
    if (Cf)  Cf [(size_t)(m0 + r) * (size_t)ldc + col] = v;
    if (Cbf) Cbf[(size_t)(m0 + r) * (size_t)ldc + col] = f2bf(v);
  }
}

// ---------------------------------------------------------------------------
// Elementwise helpers
// ---------------------------------------------------------------------------
__global__ void cvt_f32_bf16(const float* __restrict__ s,
                             unsigned short* __restrict__ d, int n) {
  int i = blockIdx.x * blockDim.x + threadIdx.x;
  if (i < n) d[i] = f2bf(s[i]);
}

__global__ void add_vec(const float* __restrict__ a, const float* __restrict__ b,
                        float* __restrict__ o, int n) {
  int i = blockIdx.x * blockDim.x + threadIdx.x;
  if (i < n) o[i] = a[i] + b[i];
}

// Build x = [E[tok], prev_ctx] as bf16  (B x (H+CTX))
__global__ void build_x(const int* __restrict__ ytok, int step,
                        const float* __restrict__ E,
                        const float* __restrict__ ctx,
                        unsigned short* __restrict__ x) {
  int i = blockIdx.x * blockDim.x + threadIdx.x;
  if (i >= B_ * HC_) return;
  int b = i >> 10, j = i & (HC_ - 1);
  int tok = (step == 0) ? 0 : ytok[b * L_ + step];
  float v = (j < H_) ? E[(size_t)tok * H_ + j] : ctx[b * CTX_ + (j - H_)];
  x[i] = f2bf(v);
}

// Zero-state LSTM cell activation: h = sigmoid(o) * tanh( sigmoid(i)*tanh(g) )
__global__ void lstm_act(const float* __restrict__ gates,   // B x 4H
                         unsigned short* __restrict__ hbf, int ldh) {
  int i = blockIdx.x * blockDim.x + threadIdx.x;
  if (i >= B_ * H_) return;
  int b = i >> 9, j = i & (H_ - 1);
  const float* g = gates + (size_t)b * G_;
  float ig = g[j], gg = g[2 * H_ + j], og = g[3 * H_ + j];
  float si = 1.0f / (1.0f + __expf(-ig));
  float so = 1.0f / (1.0f + __expf(-og));
  float c  = si * tanhf(gg);
  float h  = so * tanhf(c);
  hbf[(size_t)b * ldh + j] = f2bf(h);
}

// q0 = bf16(bq) broadcast per batch row
__global__ void init_q0(const float* __restrict__ bq,
                        unsigned short* __restrict__ q) {
  int i = blockIdx.x * blockDim.x + threadIdx.x;
  if (i >= B_ * H_) return;
  q[i] = f2bf(bq[i & (H_ - 1)]);
}

// ---------------------------------------------------------------------------
// Attention: one block (256 threads) per batch element.
//   energy[t] = <K[b,t,:], q[b,:]> ; mask ; softmax(energy/sqrt(H))
//   ctx[b,:]  = sum_t w[t] * V[b,t,:]
// K/V are bf16 -> 134 MB total, resident in the 192 MB L2 across all 399 steps.
// The V accumulation double-buffers 8-row (8 KB) tiles through LDS using the
// gfx1250 async DMA path (global_load_async_to_lds_b128 + s_wait_asynccnt),
// overlapping the next tile's L2->LDS transfer with the current accumulate.
// ---------------------------------------------------------------------------
constexpr int VROWS_ = 8;                       // V rows per LDS tile (8 KB)

__global__ void attend(const unsigned short* __restrict__ Kb,
                       const unsigned short* __restrict__ Vb,
                       const unsigned short* __restrict__ qb,   // B x H bf16
                       const int* __restrict__ lens,
                       float* __restrict__ ctx_f,               // B x CTX
                       unsigned short* __restrict__ cdn_ctx,    // B x HC (cols H..)
                       int write_cdn)
{
  __shared__ float qs[H_];
  __shared__ float e[T_];
  __shared__ float red[256];
  __shared__ unsigned short vtile[2][VROWS_ * CTX_];   // 2 x 8 KB double buffer

  const int b = blockIdx.x;
  const int tid = threadIdx.x;
  const float scale = 0.04419417382415922f;     // 1/sqrt(512)

  for (int j = tid; j < H_; j += 256) qs[j] = bf2f(qb[b * H_ + j]);
  __syncthreads();

  // ---- phase 1: energies -------------------------------------------------
  const int len = lens[b];
  float lmax = -3.4e38f;
  for (int t = tid; t < T_; t += 256) {
    const unsigned short* row = Kb + ((size_t)b * T_ + t) * H_;
    __builtin_prefetch(row + 256 * (size_t)H_, 0, 1);   // next row chunk, L2 hint
    float s = 0.0f;
    const u32x4* r4 = (const u32x4*)row;
#pragma unroll 4
    for (int c = 0; c < H_ / 8; ++c) {
      u32x4 p = r4[c];
      const unsigned short* u = (const unsigned short*)&p;
#pragma unroll
      for (int k = 0; k < 8; ++k) s += bf2f(u[k]) * qs[c * 8 + k];
    }
    s = (t < len) ? s * scale : -1.0e30f;
    e[t] = s;
    lmax = fmaxf(lmax, s);
  }
  red[tid] = lmax;
  __syncthreads();
  for (int off = 128; off; off >>= 1) {
    if (tid < off) red[tid] = fmaxf(red[tid], red[tid + off]);
    __syncthreads();
  }
  const float mx = red[0];
  __syncthreads();

  float lsum = 0.0f;
  for (int t = tid; t < T_; t += 256) {
    float w = __expf(e[t] - mx);
    e[t] = w;
    lsum += w;
  }
  red[tid] = lsum;
  __syncthreads();
  for (int off = 128; off; off >>= 1) {
    if (tid < off) red[tid] += red[tid + off];
    __syncthreads();
  }
  const float inv = 1.0f / red[0];
  __syncthreads();

  // ---- phase 2: ctx = sum_t w[t] * V[b,t,:] with async LDS staging -------
  // Tile = VROWS_ rows x CTX cols bf16 = 8192 B, contiguous in global memory.
  // Each thread DMAs 16 B at byte offset tid*16 and tid*16+4096 (INST_OFFSET
  // is added to both the LDS and the global address per the async pseudocode).
  auto stage = [&](int buf, int t0) {
    const char* g = (const char*)(Vb + ((size_t)b * T_ + t0) * CTX_) + tid * 16;
    unsigned l = (unsigned)(uintptr_t)(&vtile[buf][0]) + (unsigned)(tid * 16);
    unsigned long long ga = (unsigned long long)g;
    asm volatile("global_load_async_to_lds_b128 %0, %1, off"
                 :: "v"(l), "v"(ga) : "memory");
    asm volatile("global_load_async_to_lds_b128 %0, %1, off offset:4096"
                 :: "v"(l), "v"(ga) : "memory");
  };

  const int v0 = tid;            // CTX = 512, 256 threads -> 2 columns each
  const int v1 = tid + 256;
  float acc0 = 0.0f, acc1 = 0.0f;

  stage(0, 0);                                   // prologue
  for (int t0 = 0; t0 < T_; t0 += VROWS_) {
    const int cur = (t0 / VROWS_) & 1;
    const bool more = (t0 + VROWS_) < T_;
    if (more) stage(cur ^ 1, t0 + VROWS_);       // prefetch next tile
    // wait for current buffer's 2 DMAs (in-order); next tile's 2 may remain
    if (more) asm volatile("s_wait_asynccnt 0x2" ::: "memory");
    else      asm volatile("s_wait_asynccnt 0x0" ::: "memory");
    __syncthreads();                             // all waves' portions landed
    const unsigned short* vt = vtile[cur];
#pragma unroll
    for (int r = 0; r < VROWS_; ++r) {
      const float w = e[t0 + r];
      acc0 += w * bf2f(vt[r * CTX_ + v0]);
      acc1 += w * bf2f(vt[r * CTX_ + v1]);
    }
    __syncthreads();                             // done reading before overwrite
  }

  acc0 *= inv;
  acc1 *= inv;
  ctx_f[b * CTX_ + v0] = acc0;
  ctx_f[b * CTX_ + v1] = acc1;
  if (write_cdn) {
    cdn_ctx[(size_t)b * HC_ + H_ + v0] = f2bf(acc0);
    cdn_ctx[(size_t)b * HC_ + H_ + v1] = f2bf(acc1);
  }
}

// ---------------------------------------------------------------------------
// Host-side orchestration (all launches on `stream`; graph-capture safe)
// ---------------------------------------------------------------------------
static inline int divup(int a, int b) { return (a + b - 1) / b; }

extern "C" void kernel_launch(void* const* d_in, const int* in_sizes, int n_in,
                              void* d_out, int out_size, void* d_ws, size_t ws_size,
                              hipStream_t stream) {
  (void)in_sizes; (void)n_in; (void)out_size; (void)ws_size;

  const float* seq   = (const float*)d_in[0];
  const int*   lens  = (const int*)d_in[1];
  const int*   ytok  = (const int*)d_in[2];
  const float* E     = (const float*)d_in[3];
  const float* b_cdn = (const float*)d_in[4];
  const float* Wk    = (const float*)d_in[5];
  const float* bk    = (const float*)d_in[6];
  const float* Wq    = (const float*)d_in[7];
  const float* bq    = (const float*)d_in[8];
  const float* Wv    = (const float*)d_in[9];
  const float* bv    = (const float*)d_in[10];
  const float* Wih1  = (const float*)d_in[11];
  const float* bih1  = (const float*)d_in[12];
  const float* bhh1  = (const float*)d_in[13];
  const float* Wih2  = (const float*)d_in[14];
  const float* bih2  = (const float*)d_in[15];
  const float* bhh2  = (const float*)d_in[16];
  const float* Wt    = (const float*)d_in[17];
  const float* bt    = (const float*)d_in[18];
  float* out = (float*)d_out;

  // ---- workspace carve (256B aligned) ----
  uint8_t* ws = (uint8_t*)d_ws;
  size_t off = 0;
  auto carve = [&](size_t bytes) -> void* {
    void* p = ws + off;
    off = (off + bytes + 255) & ~(size_t)255;
    return p;
  };
  unsigned short* Xbf    = (unsigned short*)carve((size_t)B_ * T_ * H_ * 2);    // 64 MB
  unsigned short* Kbf    = (unsigned short*)carve((size_t)B_ * T_ * H_ * 2);    // 64 MB
  unsigned short* Vbf    = (unsigned short*)carve((size_t)B_ * T_ * CTX_ * 2);  // 64 MB
  unsigned short* Ebf    = (unsigned short*)carve((size_t)V_ * H_ * 2);
  unsigned short* Wkbf   = (unsigned short*)carve((size_t)H_ * H_ * 2);
  unsigned short* Wvbf   = (unsigned short*)carve((size_t)CTX_ * H_ * 2);
  unsigned short* Wqbf   = (unsigned short*)carve((size_t)H_ * H_ * 2);
  unsigned short* Wih1bf = (unsigned short*)carve((size_t)G_ * HC_ * 2);
  unsigned short* Wih2bf = (unsigned short*)carve((size_t)G_ * H_ * 2);
  unsigned short* Wtbf   = (unsigned short*)carve((size_t)H_ * HC_ * 2);
  float* bias1  = (float*)carve((size_t)G_ * 4);
  float* bias2  = (float*)carve((size_t)G_ * 4);
  unsigned short* x_bf    = (unsigned short*)carve((size_t)B_ * HC_ * 2);
  float*          gates   = (float*)carve((size_t)B_ * G_ * 4);
  unsigned short* h1bf    = (unsigned short*)carve((size_t)B_ * H_ * 2);
  unsigned short* cdn_bf  = (unsigned short*)carve((size_t)B_ * HC_ * 2);
  unsigned short* q_bf    = (unsigned short*)carve((size_t)B_ * H_ * 2);
  float*          ctx_f   = (float*)carve((size_t)B_ * CTX_ * 4);
  unsigned short* transbf = (unsigned short*)carve((size_t)B_ * H_ * 2);

  const int TB = 256;
  auto cvt = [&](const float* s, unsigned short* d, int n) {
    cvt_f32_bf16<<<divup(n, TB), TB, 0, stream>>>(s, d, n);
  };
  auto gemm = [&](const unsigned short* A, int lda, const unsigned short* W, int ldw,
                  const float* bias, float* Cf, unsigned short* Cbf, int ldc,
                  int M, int N, int K, int relu) {
    int waves = (M >> 4) * (N >> 4);
    gemm_bf16_wmma<<<divup(waves * 32, TB), TB, 0, stream>>>(
        A, lda, W, ldw, bias, Cf, Cbf, ldc, M, N, K, relu);
  };

  // ---- one-time: bf16 conversions, fused biases, K/V precompute ----
  cvt(seq,  Xbf,    B_ * T_ * H_);
  cvt(E,    Ebf,    V_ * H_);
  cvt(Wk,   Wkbf,   H_ * H_);
  cvt(Wv,   Wvbf,   CTX_ * H_);
  cvt(Wq,   Wqbf,   H_ * H_);
  cvt(Wih1, Wih1bf, G_ * HC_);
  cvt(Wih2, Wih2bf, G_ * H_);
  cvt(Wt,   Wtbf,   H_ * HC_);
  add_vec<<<divup(G_, TB), TB, 0, stream>>>(bih1, bhh1, bias1, G_);
  add_vec<<<divup(G_, TB), TB, 0, stream>>>(bih2, bhh2, bias2, G_);

  // K = X @ Wk^T + bk ; Vv = X @ Wv^T + bv   (stored bf16 -> L2-resident)
  gemm(Xbf, H_, Wkbf, H_, bk, nullptr, Kbf, H_,   B_ * T_, H_,   H_, 0);
  gemm(Xbf, H_, Wvbf, H_, bv, nullptr, Vbf, CTX_, B_ * T_, CTX_, H_, 0);

  // initial context from zero query: q0 = bq
  init_q0<<<divup(B_ * H_, TB), TB, 0, stream>>>(bq, q_bf);
  attend<<<B_, TB, 0, stream>>>(Kbf, Vbf, q_bf, lens, ctx_f, cdn_bf, 0);

  // ---- 399-step teacher-forced decode ----
  for (int s = 0; s < LOUT_; ++s) {
    build_x<<<divup(B_ * HC_, TB), TB, 0, stream>>>(ytok, s, E, ctx_f, x_bf);

    // LSTM layer 1: gates = x @ Wih1^T + (bih1+bhh1)
    gemm(x_bf, HC_, Wih1bf, HC_, bias1, gates, nullptr, G_, B_, G_, HC_, 0);
    lstm_act<<<divup(B_ * H_, TB), TB, 0, stream>>>(gates, h1bf, H_);

    // LSTM layer 2: gates = h1 @ Wih2^T + (bih2+bhh2); h2 -> cdn_in[:, :H]
    gemm(h1bf, H_, Wih2bf, H_, bias2, gates, nullptr, G_, B_, G_, H_, 0);
    lstm_act<<<divup(B_ * H_, TB), TB, 0, stream>>>(gates, cdn_bf, HC_);

    // q = h2 @ Wq^T + bq   (reads h2 out of cdn_in with lda=HC)
    gemm(cdn_bf, HC_, Wqbf, H_, bq, nullptr, q_bf, H_, B_, H_, H_, 0);

    // attention -> ctx_f (next step's prev_ctx) and cdn_in[:, H:]
    attend<<<B_, TB, 0, stream>>>(Kbf, Vbf, q_bf, lens, ctx_f, cdn_bf, 1);

    // trans = relu(cdn_in @ Wt^T + bt)
    gemm(cdn_bf, HC_, Wtbf, HC_, bt, nullptr, transbf, H_, B_, H_, HC_, 1);

    // logits = trans @ E^T + b_cdn   -> d_out[:, s, :]  (ldc = 399*V)
    gemm(transbf, H_, Ebf, H_, b_cdn, out + (size_t)s * V_, nullptr,
         LOUT_ * V_, B_, V_, H_, 0);
  }
}
